// _JinaBertAttention_28587302322545
// MI455X (gfx1250) — matrix operand order repaired and checked by reading, not compiled, not run on hardware
//
#include <hip/hip_runtime.h>

// Problem constants (JinaBERT attention): B=2, S=2048, D=1024, H=16, HD=64
#define B_  2
#define S_  2048
#define D_  1024
#define H_  16
#define HD_ 64
#define MS_ (B_ * S_)          // 4096 rows for the projection GEMMs

typedef __attribute__((ext_vector_type(16))) __bf16 v16bf;
typedef __attribute__((ext_vector_type(8)))  __bf16 v8bf;
typedef __attribute__((ext_vector_type(8)))  float  v8f;
typedef unsigned int u32x4 __attribute__((ext_vector_type(4)));
typedef int          i32x4 __attribute__((ext_vector_type(4)));
typedef int          i32x8 __attribute__((ext_vector_type(8)));

#if defined(__HIP_DEVICE_COMPILE__) && __has_builtin(__builtin_amdgcn_tensor_load_to_lds)
#define HAVE_TDM 1
#else
#define HAVE_TDM 0
#endif

static __device__ __forceinline__ __bf16 f2bf(float f) {
  unsigned u = __builtin_bit_cast(unsigned, f);
  unsigned r = u + 0x7FFFu + ((u >> 16) & 1u);   // round-to-nearest-even
  unsigned short h = (unsigned short)(r >> 16);
  return __builtin_bit_cast(__bf16, h);
}

static __device__ __forceinline__ v8f zero8() {
  v8f z;
#pragma unroll
  for (int i = 0; i < 8; ++i) z[i] = 0.0f;
  return z;
}

#if HAVE_TDM
// -----------------------------------------------------------------------------
// Tensor Data Mover descriptors (D# per CDNA5 ISA), 2D bf16 tiles.
// group1 is loop-invariant; group0 (lds_addr / global_addr) changes per issue.
// -----------------------------------------------------------------------------
static __device__ __forceinline__ i32x8 tdm_g1(
    unsigned tensor_d0, unsigned tensor_d1, unsigned stride_d0,
    unsigned tile_d0, unsigned tile_d1,
    unsigned pad_interval_code, unsigned pad_amount_code) {
  unsigned long long q0 =
      (1ull << 16)                                 // data_size = 2B
    | (1ull << 20)                                 // pad_enable
    | ((unsigned long long)pad_interval_code << 22)
    | ((unsigned long long)pad_amount_code   << 25)
    | ((unsigned long long)(tensor_d0 & 0xFFFFu) << 48);
  unsigned long long q1 =
      ((unsigned long long)tensor_d0 >> 16)        // tensor_dim0[31:16]
    | ((unsigned long long)tensor_d1 << 16)        // tensor_dim1
    | ((unsigned long long)tile_d0   << 48);       // tile_dim0
  unsigned long long q2 =
      (unsigned long long)tile_d1                  // tile_dim1 (tile_dim2 = 0)
    | ((unsigned long long)stride_d0 << 32);       // tensor_dim0_stride[31:0]
  i32x8 g1;
  g1[0] = (int)q0; g1[1] = (int)(q0 >> 32);
  g1[2] = (int)q1; g1[3] = (int)(q1 >> 32);
  g1[4] = (int)q2; g1[5] = (int)(q2 >> 32);
  g1[6] = 0;       g1[7] = 0;                      // dim1_stride unused (2D)
  return g1;
}

static __device__ __forceinline__ void tdm_issue(
    unsigned lds_byte_addr, unsigned long long ga, i32x8 g1) {
  u32x4 g0;
  g0[0] = 1u;                                      // count=1 (valid), no gather
  g0[1] = lds_byte_addr;                           // bits 63:32 lds_addr
  g0[2] = (unsigned)(ga & 0xFFFFFFFFu);            // global_addr[31:0]
  g0[3] = (unsigned)((ga >> 32) & 0x01FFFFFFu)     // global_addr[56:32]
        | (2u << 30);                              // type=2 ("image")
  i32x4 gz = {0, 0, 0, 0};
#if defined(__clang_major__) && (__clang_major__ >= 23)
  i32x8 gz8 = {0, 0, 0, 0, 0, 0, 0, 0};
  __builtin_amdgcn_tensor_load_to_lds(g0, g1, gz, gz, gz8, 0);
#else
  __builtin_amdgcn_tensor_load_to_lds(g0, g1, gz, gz, 0);
#endif
}
#endif  // HAVE_TDM

// ---------------------------------------------------------------------------
// fp32 -> bf16 conversion
// ---------------------------------------------------------------------------
__global__ void __launch_bounds__(256)
k_cvt_bf16(const float* __restrict__ src, __bf16* __restrict__ dst, int n) {
  int i = blockIdx.x * 256 + threadIdx.x;
  if (i < n) dst[i] = f2bf(src[i]);
}

// ---------------------------------------------------------------------------
// WMMA GEMM with TDM-staged LDS tiles.
//   Block: 256 threads = 8 waves, block tile 128(M) x 64(N), k-step 32.
//   Wave w owns rows [w*16, w*16+16), computes 4 C fragments (16x64).
//   mode 0: bf16 row-major out (+bias)         (Q, V projections)
//   mode 1: f32 out + residual (+bias)         (O projection)
//   mode 2: bf16 scattered to Kt[B,H,HD,S]     (K projection, fused transpose)
// LDS rows padded (TDM pad fields) for 16B alignment + bank spread:
//   A rows: 32 elems (64B) + 16B pad -> stride 40 elems
//   B rows: 64 elems (128B) + 16B pad -> stride 72 elems
// ---------------------------------------------------------------------------
#define A_STRIDE 40
#define B_STRIDE 72

__global__ void __launch_bounds__(256)
k_gemm(const __bf16* __restrict__ A, const __bf16* __restrict__ W,
       const float* __restrict__ bias, __bf16* __restrict__ outb,
       float* __restrict__ outf, const float* __restrict__ residual,
       int mode, int M, int N, int K) {
  __shared__ __bf16 asmem[2][128][A_STRIDE];
  __shared__ __bf16 bsmem[2][32][B_STRIDE];

  const int tid  = threadIdx.x;
  const int lane = tid & 31;
  const int wave = tid >> 5;
  const int half = lane >> 4;
  const int l    = lane & 15;
  const int m_block = blockIdx.y * 128;
  const int n_block = blockIdx.x * 64;
  const int nsteps  = K / 32;

  v8f c[4];
#pragma unroll
  for (int f = 0; f < 4; ++f) c[f] = zero8();

#if HAVE_TDM
  const i32x8 g1A = tdm_g1(K, M, K, 32, 128, /*16DW*/3, /*4DW*/3);
  const i32x8 g1B = tdm_g1(N, K, N, 64, 32, /*32DW*/4, /*4DW*/3);
  const unsigned ldsA0 = (unsigned)(size_t)&asmem[0][0][0];
  const unsigned ldsA1 = (unsigned)(size_t)&asmem[1][0][0];
  const unsigned ldsB0 = (unsigned)(size_t)&bsmem[0][0][0];
  const unsigned ldsB1 = (unsigned)(size_t)&bsmem[1][0][0];
  unsigned long long gaA = (unsigned long long)(size_t)(A + (size_t)m_block * K);
  unsigned long long gaB = (unsigned long long)(size_t)(W + n_block);
  const unsigned long long stepB = (unsigned long long)32 * N * 2;  // bytes per k-step
  if (wave == 0) {
    tdm_issue(ldsA0, gaA, g1A);
    tdm_issue(ldsB0, gaB, g1B);
  }
#endif

  for (int it = 0; it < nsteps; ++it) {
    const int cur = it & 1;
#if HAVE_TDM
    if (wave == 0) __builtin_amdgcn_s_wait_tensorcnt((short)0);
    __syncthreads();                       // tiles[cur] ready; tiles[1-cur] free
    if (wave == 0 && it + 1 < nsteps) {
      tdm_issue(cur ? ldsA0 : ldsA1, gaA + 64ull,  g1A);  // +32 bf16 along K
      tdm_issue(cur ? ldsB0 : ldsB1, gaB + stepB, g1B);   // +32 rows
    }
    gaA += 64ull;
    gaB += stepB;
#else
    const int k0 = it * 32;
    __syncthreads();
    for (int e = tid; e < 128 * 32; e += 256) {
      int r = e >> 5, cc = e & 31;
      asmem[cur][r][cc] = A[(size_t)(m_block + r) * K + k0 + cc];
    }
    for (int e = tid; e < 32 * 64; e += 256) {
      int r = e >> 6, cc = e & 63;
      bsmem[cur][r][cc] = W[(size_t)(k0 + r) * N + n_block + cc];
    }
    __syncthreads();
#endif
    // A fragment: lane row = wave*16 + l; K = half*8+{0..7}, 16+half*8+{0..7}
    const __bf16* arow = &asmem[cur][wave * 16 + l][0];
    v8bf alo = *(const v8bf*)(arow + half * 8);
    v8bf ahi = *(const v8bf*)(arow + 16 + half * 8);
    v16bf afrag;
#pragma unroll
    for (int e = 0; e < 8; ++e) { afrag[e] = alo[e]; afrag[8 + e] = ahi[e]; }
    // batch all 4 B-fragment LDS loads, then 4 WMMAs (amortize s_wait_dscnt)
    v16bf bfr[4];
#pragma unroll
    for (int f = 0; f < 4; ++f)
#pragma unroll
      for (int s = 0; s < 16; ++s)
        bfr[f][s] = bsmem[cur][(s & 7) + half * 8 + (s >> 3) * 16][f * 16 + l];
#pragma unroll
    for (int f = 0; f < 4; ++f)
      c[f] = __builtin_amdgcn_wmma_f32_16x16x32_bf16(false, afrag, false, bfr[f],
                                                     (short)0, c[f], false, false);
  }

  // epilogue
  const int m_base = m_block + wave * 16;
#pragma unroll
  for (int j = 0; j < 8; ++j) {
    const int r = m_base + j + 8 * half;       // C layout: VGPR j -> row j+8*half
#pragma unroll
    for (int f = 0; f < 4; ++f) {
      const int n = n_block + f * 16 + l;
      float v = c[f][j] + bias[n];
      if (mode == 0) {
        outb[(size_t)r * N + n] = f2bf(v);
      } else if (mode == 1) {
        size_t idx = (size_t)r * N + n;
        outf[idx] = residual[idx] + v;
      } else {  // mode 2: Kt[B,H,HD,S] fused transpose
        int b  = r >> 11, s = r & (S_ - 1);
        int h  = n >> 6,  hd = n & (HD_ - 1);
        outb[(((size_t)(b * H_ + h)) * HD_ + hd) * S_ + s] = f2bf(v);
      }
    }
  }
}

// ---------------------------------------------------------------------------
// Flash attention: 4 waves/block, one (b, h, 16-query tile) per wave.
// scores = Q@K^T * scale + mask + alibi ; online softmax ; O += P@V
// ---------------------------------------------------------------------------
__global__ void __launch_bounds__(128)
k_attn(const __bf16* __restrict__ qbf, const __bf16* __restrict__ ktbf,
       const __bf16* __restrict__ vbf, const float* __restrict__ amask,
       const float* __restrict__ alibi, __bf16* __restrict__ obf) {
  const int lane = threadIdx.x & 31;
  const int wave = threadIdx.x >> 5;
  const int half = lane >> 4;
  const int l    = lane & 15;
  const int q0 = (blockIdx.x * 4 + wave) * 16;
  const int h  = blockIdx.y;
  const int b  = blockIdx.z;
  const float scale = 0.125f;       // HD^-0.5 = 1/8

  __shared__ __bf16 pbuf[4][16][32];   // per-wave P staging (C-layout -> A-layout)

  // Q tile 16x64 as two 16x32 A fragments
  const __bf16* qhead = qbf + (size_t)b * S_ * D_ + (size_t)h * HD_;
  v16bf aq[2];
#pragma unroll
  for (int t = 0; t < 2; ++t) {
    const __bf16* row = qhead + (size_t)(q0 + l) * D_ + t * 32 + half * 8;
    v8bf lo = *(const v8bf*)(row);
    v8bf hi = *(const v8bf*)(row + 16);
#pragma unroll
    for (int e = 0; e < 8; ++e) { aq[t][e] = lo[e]; aq[t][8 + e] = hi[e]; }
  }

  const __bf16* kthead = ktbf + ((size_t)(b * H_ + h)) * HD_ * S_;
  const __bf16* vhead  = vbf + (size_t)b * S_ * D_ + (size_t)h * HD_;
  const float*  mrow   = amask + (size_t)b * S_;
  const float*  arow   = alibi + (size_t)h * S_ * S_;

  v8f oacc[4];
#pragma unroll
  for (int f = 0; f < 4; ++f) oacc[f] = zero8();
  float mrun[8], lrun[8];
#pragma unroll
  for (int j = 0; j < 8; ++j) { mrun[j] = -3.0e38f; lrun[j] = 0.0f; }

  for (int kb = 0; kb < S_; kb += 32) {
    if (kb + 32 < S_)  // warm L2 for the next alibi tile (global_prefetch_b8)
      __builtin_prefetch(&arow[(size_t)(q0 + l) * S_ + kb + 32], 0, 0);

    v8f sc[2];
#pragma unroll
    for (int t = 0; t < 2; ++t) {
      int n0 = kb + t * 16;
      v16bf bk0, bk1;
#pragma unroll
      for (int s = 0; s < 16; ++s) {
        int kk = (s & 7) + half * 8 + (s >> 3) * 16;
        bk0[s] = kthead[(size_t)kk * S_ + n0 + l];
        bk1[s] = kthead[(size_t)(32 + kk) * S_ + n0 + l];
      }
      v8f s = zero8();
      s = __builtin_amdgcn_wmma_f32_16x16x32_bf16(false, aq[0], false, bk0,
                                                  (short)0, s, false, false);
      s = __builtin_amdgcn_wmma_f32_16x16x32_bf16(false, aq[1], false, bk1,
                                                  (short)0, s, false, false);
#pragma unroll
      for (int j = 0; j < 8; ++j) {
        int r = q0 + j + 8 * half;
        int n = n0 + l;
        s[j] = s[j] * scale + mrow[n] + arow[(size_t)r * S_ + n];
      }
      sc[t] = s;
    }

    // online softmax (row = j + 8*half, replicated across the 16 lanes of a half)
#pragma unroll
    for (int j = 0; j < 8; ++j) {
      float rm = fmaxf(sc[0][j], sc[1][j]);
#pragma unroll
      for (int o = 1; o <= 8; o <<= 1) rm = fmaxf(rm, __shfl_xor(rm, o, 32));
      float mnew  = fmaxf(mrun[j], rm);
      float alpha = __expf(mrun[j] - mnew);
      float p0 = __expf(sc[0][j] - mnew);
      float p1 = __expf(sc[1][j] - mnew);
      float rs = p0 + p1;
#pragma unroll
      for (int o = 1; o <= 8; o <<= 1) rs += __shfl_xor(rs, o, 32);
      lrun[j] = lrun[j] * alpha + rs;
      mrun[j] = mnew;
#pragma unroll
      for (int f = 0; f < 4; ++f) oacc[f][j] *= alpha;
      sc[0][j] = p0; sc[1][j] = p1;
    }

    // stage P (16x32) into LDS row-major, reload in A-fragment layout
#pragma unroll
    for (int j = 0; j < 8; ++j) {
      int r = j + 8 * half;
      pbuf[wave][r][l]      = f2bf(sc[0][j]);
      pbuf[wave][r][16 + l] = f2bf(sc[1][j]);
    }
    asm volatile("s_wait_dscnt 0" ::: "memory");
    v8bf plo = *(const v8bf*)(&pbuf[wave][l][half * 8]);
    v8bf phi = *(const v8bf*)(&pbuf[wave][l][16 + half * 8]);
    v16bf ap;
#pragma unroll
    for (int e = 0; e < 8; ++e) { ap[e] = plo[e]; ap[8 + e] = phi[e]; }

    // O += P @ V   (V rows kb..kb+31, four 16-wide HD column tiles)
#pragma unroll
    for (int f = 0; f < 4; ++f) {
      v16bf bv;
#pragma unroll
      for (int s = 0; s < 16; ++s)
        bv[s] = vhead[(size_t)(kb + (s & 7) + half * 8 + (s >> 3) * 16) * D_ + f * 16 + l];
      oacc[f] = __builtin_amdgcn_wmma_f32_16x16x32_bf16(false, ap, false, bv,
                                                        (short)0, oacc[f], false, false);
    }
  }

  // normalize and store attn output (bf16, [B*S, D] head-interleaved)
#pragma unroll
  for (int j = 0; j < 8; ++j) {
    float inv = 1.0f / lrun[j];
    int r = q0 + j + 8 * half;
#pragma unroll
    for (int f = 0; f < 4; ++f) {
      obf[((size_t)(b * S_ + r)) * D_ + h * HD_ + f * 16 + l] =
          f2bf(oacc[f][j] * inv);
    }
  }
}

// ---------------------------------------------------------------------------
// Row LayerNorm over D=1024: one block (256 threads) per row.
// ---------------------------------------------------------------------------
__global__ void __launch_bounds__(256)
k_layernorm(const float* __restrict__ x, const float* __restrict__ gamma,
            const float* __restrict__ beta, float* __restrict__ out) {
  const int row = blockIdx.x;
  const float* xr = x + (size_t)row * D_;
  __shared__ float reds[8], reds2[8];
  float v4[4];
  float s = 0.0f, s2 = 0.0f;
#pragma unroll
  for (int i = 0; i < 4; ++i) {
    float v = xr[threadIdx.x + i * 256];
    v4[i] = v; s += v; s2 += v * v;
  }
#pragma unroll
  for (int o = 1; o < 32; o <<= 1) {
    s  += __shfl_xor(s,  o, 32);
    s2 += __shfl_xor(s2, o, 32);
  }
  int wave = threadIdx.x >> 5;
  if ((threadIdx.x & 31) == 0) { reds[wave] = s; reds2[wave] = s2; }
  __syncthreads();
  s = 0.0f; s2 = 0.0f;
#pragma unroll
  for (int i = 0; i < 8; ++i) { s += reds[i]; s2 += reds2[i]; }
  float mu  = s * (1.0f / D_);
  float var = s2 * (1.0f / D_) - mu * mu;
  float inv = rsqrtf(var + 1e-12f);
#pragma unroll
  for (int i = 0; i < 4; ++i) {
    int col = threadIdx.x + i * 256;
    out[(size_t)row * D_ + col] = (v4[i] - mu) * inv * gamma[col] + beta[col];
  }
}

// ---------------------------------------------------------------------------
extern "C" void kernel_launch(void* const* d_in, const int* in_sizes, int n_in,
                              void* d_out, int out_size, void* d_ws, size_t ws_size,
                              hipStream_t stream) {
  const float* hidden = (const float*)d_in[0];
  const float* amask  = (const float*)d_in[1];
  const float* alibi  = (const float*)d_in[2];
  const float* Wq = (const float*)d_in[3];
  const float* bq = (const float*)d_in[4];
  const float* Wk = (const float*)d_in[5];
  const float* bk = (const float*)d_in[6];
  const float* Wv = (const float*)d_in[7];
  const float* bv = (const float*)d_in[8];
  const float* Wo = (const float*)d_in[9];
  const float* bo = (const float*)d_in[10];
  const float* gamma = (const float*)d_in[11];
  const float* beta  = (const float*)d_in[12];
  float* out = (float*)d_out;

  const size_t MK = (size_t)MS_ * D_;   // 4096*1024
  const size_t DD = (size_t)D_ * D_;    // 1024*1024

  char* ws = (char*)d_ws;
  size_t off = 0;
  auto alloc = [&](size_t bytes) -> void* {
    void* p = ws + off;
    off = (off + bytes + 255) & ~(size_t)255;
    return p;
  };
  __bf16* hs_bf = (__bf16*)alloc(MK * 2);
  __bf16* wq_bf = (__bf16*)alloc(DD * 2);
  __bf16* wk_bf = (__bf16*)alloc(DD * 2);
  __bf16* wv_bf = (__bf16*)alloc(DD * 2);
  __bf16* wo_bf = (__bf16*)alloc(DD * 2);
  __bf16* q_bf  = (__bf16*)alloc(MK * 2);
  __bf16* kt_bf = (__bf16*)alloc(MK * 2);
  __bf16* v_bf  = (__bf16*)alloc(MK * 2);
  __bf16* a_bf  = (__bf16*)alloc(MK * 2);
  float*  x_f32 = (float*) alloc(MK * 4);
  (void)ws_size; (void)in_sizes; (void)n_in; (void)out_size;

  // 1) bf16 conversions
  k_cvt_bf16<<<(int)((MK + 255) / 256), 256, 0, stream>>>(hidden, hs_bf, (int)MK);
  k_cvt_bf16<<<(int)((DD + 255) / 256), 256, 0, stream>>>(Wq, wq_bf, (int)DD);
  k_cvt_bf16<<<(int)((DD + 255) / 256), 256, 0, stream>>>(Wk, wk_bf, (int)DD);
  k_cvt_bf16<<<(int)((DD + 255) / 256), 256, 0, stream>>>(Wv, wv_bf, (int)DD);
  k_cvt_bf16<<<(int)((DD + 255) / 256), 256, 0, stream>>>(Wo, wo_bf, (int)DD);

  // 2) QKV projections (WMMA bf16, TDM-staged LDS tiles)
  dim3 ggrid(D_ / 64, MS_ / 128);
  k_gemm<<<ggrid, 256, 0, stream>>>(hs_bf, wq_bf, bq, q_bf,  nullptr, nullptr, 0, MS_, D_, D_);
  k_gemm<<<ggrid, 256, 0, stream>>>(hs_bf, wk_bf, bk, kt_bf, nullptr, nullptr, 2, MS_, D_, D_);
  k_gemm<<<ggrid, 256, 0, stream>>>(hs_bf, wv_bf, bv, v_bf,  nullptr, nullptr, 0, MS_, D_, D_);

  // 3) flash attention (WMMA bf16, online softmax); alibi (268 MB) read once
  k_attn<<<dim3(S_ / 64, H_, B_), 128, 0, stream>>>(q_bf, kt_bf, v_bf, amask, alibi, a_bf);

  // 4) output projection + residual (WMMA bf16, f32 accum/out)
  k_gemm<<<ggrid, 256, 0, stream>>>(a_bf, wo_bf, bo, nullptr, x_f32, hidden, 1, MS_, D_, D_);

  // 5) LayerNorm
  k_layernorm<<<MS_, 256, 0, stream>>>(x_f32, gamma, beta, out);
}
